// NDLNormAttention_51788715655576
// MI455X (gfx1250) — matrix operand order repaired and checked by reading, not compiled, not run on hardware
//
#include <hip/hip_runtime.h>
#include <hip/hip_bf16.h>

#define HIDDEN  2048
#define SEQ     2048
#define NHEADS  32
#define KVHEADS 8
#define HEADDIM 64
#define NQ      (NHEADS * HEADDIM)   // 2048
#define NKV     (KVHEADS * HEADDIM)  // 512

typedef __attribute__((ext_vector_type(16))) __bf16 v16bf;
typedef __attribute__((ext_vector_type(8)))  __bf16 v8bf;
typedef __attribute__((ext_vector_type(8)))  float  v8f;

// Load a 16x32 bf16 A-fragment for one lane.
// Layout used here (contiguous k): lane = half*16 + m (m = row), element e -> K = half*16+e.
// "base" points at row m, column k0 of a row-major [M][K] matrix (global or LDS).
__device__ __forceinline__ v16bf load_a_frag(const __bf16* __restrict__ base, int half) {
  v8bf lo = *(const v8bf*)(base + half * 16);
  v8bf hi = *(const v8bf*)(base + half * 16 + 8);
  v16bf a;
#pragma unroll
  for (int i = 0; i < 8; ++i) { a[i] = lo[i]; a[i + 8] = hi[i]; }
  return a;
}

// Per-lane async DMA: 16B from global -> LDS, tracked by ASYNCcnt (CDNA5 path).
__device__ __forceinline__ void async_ld16(unsigned lds_off, unsigned long long gaddr) {
  asm volatile("global_load_async_to_lds_b128 %0, %1, off"
               :: "v"(lds_off), "v"(gaddr) : "memory");
}
__device__ __forceinline__ void wait_async0() {
  asm volatile("s_wait_asynccnt 0" ::: "memory");
}

// ---------------------------------------------------------------------------
// GEMM: C[M][N] (f32) = A[M][K] (bf16, row-major) @ Bt[N][K] (bf16 = B^T).
// 256 threads = 8 waves; block tile 128x128, each wave an independent 64x32
// tile: 4 A-frags x 2 B-frags -> 8 x v_wmma_f32_16x16x32_bf16 per 32-wide K
// step (21 FLOP per loaded byte; operands are L2-resident on MI455X's 192MB L2).
// ---------------------------------------------------------------------------
__global__ void gemm_bf16_wmma(const __bf16* __restrict__ A,
                               const __bf16* __restrict__ Bt,
                               float* __restrict__ C,
                               int M, int N, int K) {
  const int lane = threadIdx.x & 31;
  const int wave = threadIdx.x >> 5;     // 0..7
  const int half = lane >> 4;
  const int l16  = lane & 15;
  const int tilesN = N >> 7;
  const int bm = (blockIdx.x / tilesN) << 7;
  const int bn = (blockIdx.x % tilesN) << 7;
  const int tm = bm + ((wave >> 2) << 6);   // 2 wave-rows of 64
  const int tn = bn + ((wave & 3) << 5);    // 4 wave-cols of 32

  v8f acc[4][2];
#pragma unroll
  for (int i = 0; i < 4; ++i)
#pragma unroll
    for (int j = 0; j < 2; ++j) acc[i][j] = (v8f){};

  const __bf16* ap[4];
  const __bf16* bp[2];
#pragma unroll
  for (int i = 0; i < 4; ++i) ap[i] = A + (size_t)(tm + i * 16 + l16) * K;
#pragma unroll
  for (int j = 0; j < 2; ++j) bp[j] = Bt + (size_t)(tn + j * 16 + l16) * K + half * 16;

  for (int k0 = 0; k0 < K; k0 += 32) {
    v16bf a[4], b[2];
#pragma unroll
    for (int i = 0; i < 4; ++i) a[i] = load_a_frag(ap[i] + k0, half);
#pragma unroll
    for (int j = 0; j < 2; ++j) b[j] = *(const v16bf*)(bp[j] + k0);
#pragma unroll
    for (int i = 0; i < 4; ++i)
#pragma unroll
      for (int j = 0; j < 2; ++j)
        acc[i][j] = __builtin_amdgcn_wmma_f32_16x16x32_bf16(
            false, a[i], false, b[j], (short)0, acc[i][j], false, false);
  }
  // C/D layout: lane holds column n = l16; VGPR r holds row m = r + 8*half.
#pragma unroll
  for (int i = 0; i < 4; ++i)
#pragma unroll
    for (int r = 0; r < 8; ++r) {
      const int m0 = tm + i * 16 + r + 8 * half;
      C[(size_t)m0 * N + tn + l16]      = acc[i][0][r];
      C[(size_t)m0 * N + tn + 16 + l16] = acc[i][1][r];
    }
}

// ---------------------------------------------------------------------------
// Flash attention, causal, GQA (4 Q heads per KV head).
// Block = 128 threads = 4 waves; blockIdx = (q_block_of_64, head).
// Each wave owns one 16-row Q tile (Q frags stay in registers) and iterates
// 32-column causal KV chunks. K/V tiles are staged into per-wave LDS slices
// with double-buffered GLOBAL_LOAD_ASYNC_TO_LDS_B128 (ASYNCcnt), hiding load
// latency behind the serial WMMA->softmax->WMMA chain with zero VGPR cost.
// All staging is per-wave: no cross-wave barriers, so per-wave trip counts
// are safe and EXEC stays all-ones around every WMMA.
// ---------------------------------------------------------------------------
__global__ void flash_attn_wmma(const __bf16* __restrict__ Qb,
                                const __bf16* __restrict__ Kb,
                                const __bf16* __restrict__ Vt,
                                __bf16* __restrict__ Cb) {
  const int lane = threadIdx.x & 31;
  const int wave = threadIdx.x >> 5;      // 0..3
  const int half = lane >> 4;
  const int l16  = lane & 15;
  const int h    = blockIdx.y;
  const int kvh  = h >> 2;                // N_REP = 4
  const int qt   = (blockIdx.x << 6) + (wave << 4);

  __shared__ __bf16 Ks[4][2][32][64];     // per-wave double-buffered K tile (32 kv x 64 d)
  __shared__ __bf16 Vs[4][2][64][32];     // per-wave double-buffered V tile (64 d x 32 kv)
  __shared__ __bf16 Pl[4][16][32];        // per-wave P staging (C-layout -> A-layout)

  // Generic shared-aperture addresses carry the LDS byte offset in addr[31:0].
  const unsigned ksBase = (unsigned)(size_t)&Ks[wave][0][0][0];
  const unsigned vsBase = (unsigned)(size_t)&Vs[wave][0][0][0];

  const __bf16* qrow = Qb + (size_t)(qt + l16) * NQ + h * HEADDIM;
  const v16bf qa0 = load_a_frag(qrow,      half);   // head-dim 0..31
  const v16bf qa1 = load_a_frag(qrow + 32, half);   // head-dim 32..63

  v8f o0 = {}, o1 = {}, o2 = {}, o3 = {};
  float rm[8], rl[8];
#pragma unroll
  for (int r = 0; r < 8; ++r) { rm[r] = -3.0e38f; rl[r] = 0.0f; }

  const int nchunks = (qt + 16 + 31) >> 5;          // uniform per wave

  // Issue async staging of chunk `base` into buffer `b` (16 x 512B DMA ops).
  auto stage = [&](int base, int b) {
#pragma unroll
    for (int s = 0; s < 8; ++s) {          // K tile: 32 rows x 128B = 256 x 16B segs
      int seg = s * 32 + lane;
      int row = seg >> 3, c16 = seg & 7;
      async_ld16(ksBase + (unsigned)b * 4096u + (unsigned)seg * 16u,
                 (unsigned long long)(size_t)(Kb + (size_t)(base + row) * NKV +
                                              kvh * HEADDIM + c16 * 8));
    }
#pragma unroll
    for (int s = 0; s < 8; ++s) {          // V tile: 64 rows x 64B = 256 x 16B segs
      int seg = s * 32 + lane;
      int row = seg >> 2, c16 = seg & 3;
      async_ld16(vsBase + (unsigned)b * 4096u + (unsigned)seg * 16u,
                 (unsigned long long)(size_t)(Vt + (size_t)(kvh * HEADDIM + row) * SEQ +
                                              base + c16 * 8));
    }
  };

  stage(0, 0);
  for (int c = 0; c < nchunks; ++c) {
    const int base = c << 5;
    const int buf  = c & 1;
    wait_async0();                               // chunk c resident in LDS
    if (c + 1 < nchunks) stage(base + 32, buf ^ 1);  // prefetch next chunk

    const __bf16* Kc = &Ks[wave][buf][0][0];
    const __bf16* Vc = &Vs[wave][buf][0][0];

    v8f sc[2];
#pragma unroll
    for (int t = 0; t < 2; ++t) {
      // K^T B-fragment from LDS: lane col = kv row t*16+l16, element e -> d = half*16+e.
      const __bf16* kp = Kc + (size_t)(t * 16 + l16) * 64 + half * 16;
      v16bf kb0 = *(const v16bf*)(kp);       // d = 0..31 pattern
      v16bf kb1 = *(const v16bf*)(kp + 32);  // d = 32..63 pattern
      v8f acc = {};
      acc = __builtin_amdgcn_wmma_f32_16x16x32_bf16(false, qa0, false, kb0, (short)0, acc, false, false);
      acc = __builtin_amdgcn_wmma_f32_16x16x32_bf16(false, qa1, false, kb1, (short)0, acc, false, false);
      sc[t] = acc;
    }
    // online softmax per row (rows of this lane: m = qt + r + 8*half)
#pragma unroll
    for (int r = 0; r < 8; ++r) {
      const int m = qt + r + 8 * half;
      float s0 = (base + l16      <= m) ? sc[0][r] : -3.0e38f;
      float s1 = (base + 16 + l16 <= m) ? sc[1][r] : -3.0e38f;
      float mx = fmaxf(s0, s1);
#pragma unroll
      for (int off = 1; off < 16; off <<= 1) mx = fmaxf(mx, __shfl_xor(mx, off, 32));
      const float nm    = fmaxf(rm[r], mx);
      const float alpha = __expf(rm[r] - nm);
      const float p0    = __expf(s0 - nm);
      const float p1    = __expf(s1 - nm);
      float rs = p0 + p1;
#pragma unroll
      for (int off = 1; off < 16; off <<= 1) rs += __shfl_xor(rs, off, 32);
      rl[r] = rl[r] * alpha + rs;
      rm[r] = nm;
      o0[r] *= alpha; o1[r] *= alpha; o2[r] *= alpha; o3[r] *= alpha;
      Pl[wave][r + 8 * half][l16]      = (__bf16)p0;
      Pl[wave][r + 8 * half][16 + l16] = (__bf16)p1;
    }
    // P (16x32) as A-fragment from LDS (same-wave RAW: compiler inserts dscnt wait)
    const v16bf pa = load_a_frag(&Pl[wave][l16][0], half);
    // V B-fragments from LDS: lane col d = dt*16+l16, element e -> kv = half*16+e.
    {
      const __bf16* vp = Vc + (size_t)l16 * 32 + half * 16;
      v16bf vb0 = *(const v16bf*)(vp);
      v16bf vb1 = *(const v16bf*)(vp + 16 * 32);
      v16bf vb2 = *(const v16bf*)(vp + 32 * 32);
      v16bf vb3 = *(const v16bf*)(vp + 48 * 32);
      o0 = __builtin_amdgcn_wmma_f32_16x16x32_bf16(false, pa, false, vb0, (short)0, o0, false, false);
      o1 = __builtin_amdgcn_wmma_f32_16x16x32_bf16(false, pa, false, vb1, (short)0, o1, false, false);
      o2 = __builtin_amdgcn_wmma_f32_16x16x32_bf16(false, pa, false, vb2, (short)0, o2, false, false);
      o3 = __builtin_amdgcn_wmma_f32_16x16x32_bf16(false, pa, false, vb3, (short)0, o3, false, false);
    }
  }
  // normalize and emit context in [s][h*64+d] layout (bf16 for the O-proj GEMM)
#pragma unroll
  for (int r = 0; r < 8; ++r) {
    const float inv = 1.0f / rl[r];
    __bf16* co = Cb + (size_t)(qt + r + 8 * half) * HIDDEN + h * HEADDIM;
    co[l16]      = (__bf16)(o0[r] * inv);
    co[16 + l16] = (__bf16)(o1[r] * inv);
    co[32 + l16] = (__bf16)(o2[r] * inv);
    co[48 + l16] = (__bf16)(o3[r] * inv);
  }
}

// ---------------------------------------------------------------------------
// Elementwise / layout kernels
// ---------------------------------------------------------------------------
__global__ void cvt_f32_bf16(const float* __restrict__ X, __bf16* __restrict__ Y, int n) {
  int i = blockIdx.x * blockDim.x + threadIdx.x;
  if (i < n) Y[i] = (__bf16)X[i];
}

// W[K][N] (f32, row-major) -> Wt[N][K] (bf16)
__global__ void transpose_cvt(const float* __restrict__ W, __bf16* __restrict__ Wt, int K, int N) {
  int idx = blockIdx.x * blockDim.x + threadIdx.x;
  if (idx >= K * N) return;
  int n = idx / K, k = idx % K;
  Wt[(size_t)n * K + k] = (__bf16)W[(size_t)k * N + n];
}

// RoPE + optional 1/sqrt(head_dim) scale, f32 in -> bf16 out; one thread per (s, head, pair)
__global__ void rope_bf16(const float* __restrict__ X, const int* __restrict__ pos,
                          __bf16* __restrict__ Y, int nheads, float scale) {
  const int stride = nheads * HEADDIM;
  int idx = blockIdx.x * blockDim.x + threadIdx.x;
  int i  = idx & 31;
  int t  = idx >> 5;
  int hh = t % nheads;
  int s  = t / nheads;
  if (s >= SEQ) return;
  float p = (float)pos[s];
  float inv_freq = __powf(10000.0f, -(float)i * (1.0f / 32.0f));  // theta^(-2i/64)
  float ang = p * inv_freq;
  float cs = __cosf(ang), sn = __sinf(ang);
  const float* x = X + (size_t)s * stride + hh * HEADDIM;
  __bf16*      y = Y + (size_t)s * stride + hh * HEADDIM;
  float x1 = x[i], x2 = x[i + 32];
  y[i]      = (__bf16)((x1 * cs - x2 * sn) * scale);
  y[i + 32] = (__bf16)((x2 * cs + x1 * sn) * scale);
}

// Vf[s][kvh*64+d] (f32) -> Vt[kvh*64+d][s] (bf16)
__global__ void vtrans_bf16(const float* __restrict__ Vf, __bf16* __restrict__ Vt) {
  int idx = blockIdx.x * blockDim.x + threadIdx.x;
  if (idx >= SEQ * NKV) return;
  int cidx = idx % NKV;
  int s    = idx / NKV;
  Vt[(size_t)cidx * SEQ + s] = (__bf16)Vf[(size_t)s * NKV + cidx];
}

// ---------------------------------------------------------------------------
extern "C" void kernel_launch(void* const* d_in, const int* in_sizes, int n_in,
                              void* d_out, int out_size, void* d_ws, size_t ws_size,
                              hipStream_t stream) {
  (void)in_sizes; (void)n_in; (void)out_size; (void)ws_size;
  const float* hid = (const float*)d_in[0];
  const float* Wq  = (const float*)d_in[1];
  const float* Wk  = (const float*)d_in[2];
  const float* Wv  = (const float*)d_in[3];
  const float* Wo  = (const float*)d_in[4];
  const int*   pos = (const int*)d_in[5];

  char* ws = (char*)d_ws;
  size_t off = 0;
  auto alloc = [&](size_t bytes) -> void* {
    void* p = ws + off;
    off += (bytes + 255) & ~(size_t)255;
    return p;
  };
  __bf16* Xb  = (__bf16*)alloc((size_t)SEQ * HIDDEN * 2);   // 8 MB
  __bf16* Wqt = (__bf16*)alloc((size_t)NQ  * HIDDEN * 2);   // 8 MB
  __bf16* Wkt = (__bf16*)alloc((size_t)NKV * HIDDEN * 2);   // 2 MB
  __bf16* Wvt = (__bf16*)alloc((size_t)NKV * HIDDEN * 2);   // 2 MB
  __bf16* Wot = (__bf16*)alloc((size_t)HIDDEN * HIDDEN * 2);// 8 MB
  float*  Qf  = (float*) alloc((size_t)SEQ * NQ  * 4);      // 16 MB
  float*  Kf  = (float*) alloc((size_t)SEQ * NKV * 4);      // 4 MB
  float*  Vf  = (float*) alloc((size_t)SEQ * NKV * 4);      // 4 MB
  __bf16* Qb  = (__bf16*)alloc((size_t)SEQ * NQ  * 2);      // 8 MB (roped, pre-scaled)
  __bf16* Kb  = (__bf16*)alloc((size_t)SEQ * NKV * 2);      // 2 MB (roped)
  __bf16* Vt  = (__bf16*)alloc((size_t)NKV * SEQ * 2);      // 2 MB (transposed)
  __bf16* Cb  = (__bf16*)alloc((size_t)SEQ * HIDDEN * 2);   // 8 MB (context)

  const int T = 256;
  cvt_f32_bf16<<<(SEQ * HIDDEN + T - 1) / T, T, 0, stream>>>(hid, Xb, SEQ * HIDDEN);
  transpose_cvt<<<(HIDDEN * NQ  + T - 1) / T, T, 0, stream>>>(Wq, Wqt, HIDDEN, NQ);
  transpose_cvt<<<(HIDDEN * NKV + T - 1) / T, T, 0, stream>>>(Wk, Wkt, HIDDEN, NKV);
  transpose_cvt<<<(HIDDEN * NKV + T - 1) / T, T, 0, stream>>>(Wv, Wvt, HIDDEN, NKV);
  transpose_cvt<<<(HIDDEN * HIDDEN + T - 1) / T, T, 0, stream>>>(Wo, Wot, HIDDEN, HIDDEN);

  // QKV projections (WMMA, 128x128 block tiles / 64x32 wave tiles)
  gemm_bf16_wmma<<<(SEQ / 128) * (NQ  / 128), 256, 0, stream>>>(Xb, Wqt, Qf, SEQ, NQ,  HIDDEN);
  gemm_bf16_wmma<<<(SEQ / 128) * (NKV / 128), 256, 0, stream>>>(Xb, Wkt, Kf, SEQ, NKV, HIDDEN);
  gemm_bf16_wmma<<<(SEQ / 128) * (NKV / 128), 256, 0, stream>>>(Xb, Wvt, Vf, SEQ, NKV, HIDDEN);

  // RoPE (fold 1/sqrt(64) into Q), V transpose
  rope_bf16<<<(SEQ * NHEADS  * 32 + T - 1) / T, T, 0, stream>>>(Qf, pos, Qb, NHEADS,  0.125f);
  rope_bf16<<<(SEQ * KVHEADS * 32 + T - 1) / T, T, 0, stream>>>(Kf, pos, Kb, KVHEADS, 1.0f);
  vtrans_bf16<<<(SEQ * NKV + T - 1) / T, T, 0, stream>>>(Vf, Vt);

  // Causal flash attention (WMMA + async-to-LDS double buffering)
  flash_attn_wmma<<<dim3(SEQ / 64, NHEADS), 128, 0, stream>>>(Qb, Kb, Vt, Cb);

  // Output projection straight to f32 d_out (WMMA)
  gemm_bf16_wmma<<<(SEQ / 128) * (HIDDEN / 128), 256, 0, stream>>>(Cb, Wot, (float*)d_out, SEQ, HIDDEN, HIDDEN);
}
